// FrozenRepAdapterHierarchical_74363063763561
// MI455X (gfx1250) — matrix operand order repaired and checked by reading, not compiled, not run on hardware
//
#include <hip/hip_runtime.h>
#include <hip/hip_bf16.h>
#include <math.h>

typedef _Float16 v16h __attribute__((ext_vector_type(16)));
typedef _Float16 v8h  __attribute__((ext_vector_type(8)));
typedef float    v8f  __attribute__((ext_vector_type(8)));
typedef float    v4f  __attribute__((ext_vector_type(4)));

#define WAVES 8
#define BLOCK 256
#define DDIM  128
#define RDIM  16
#define KPAD  32   // up-proj K padded to 32 (rows 16..31 are stored zeros)

__global__ __launch_bounds__(BLOCK)
void adapter3_fused_wmma_kernel(const float* __restrict__ xin,
                                const float* __restrict__ gamma,
                                const float* __restrict__ beta,
                                const float* __restrict__ Wd,
                                const float* __restrict__ bd,
                                const float* __restrict__ Wu,
                                const float* __restrict__ bu,
                                float* __restrict__ out,
                                int nTiles, int nTileBlocks)
{
    // Weights staged once per block. Down-proj weights are pre-scaled by gamma;
    // the LN affine is folded into per-R vectors u,v so the WMMA consumes raw x.
    __shared__ __align__(16) _Float16 sWdT[3][RDIM][DDIM];  // [a][r][d] = gamma[d]*Wd[d][r]
    __shared__ __align__(16) _Float16 sWuT[3][DDIM][KPAD];  // [a][d][r], r>=16 zero
    __shared__ float sU[3][RDIM];   // u_r = sum_d gamma_d Wd[d][r]
    __shared__ float sV[3][RDIM];   // v_r = sum_d beta_d  Wd[d][r] + bd_r
    __shared__ float sBu[3][DDIM];
    // Per-wave working tiles (each wave only touches its own slice).
    __shared__ __align__(16) float    xbuf[WAVES][16][DDIM];  // running residual, f32
    __shared__ __align__(16) _Float16 hbuf[WAVES][16][DDIM];  // f16(x), WMMA A feed
    __shared__ __align__(16) _Float16 gbuf[WAVES][16][RDIM];  // GELU output, f16
    __shared__ __align__(16) float2   sStats[WAVES][16];      // per token: (rs, -mu*rs)

    const int tid  = threadIdx.x;
    const int wid  = tid >> 5;
    const int lane = tid & 31;
    const int t    = lane & 15;   // token-in-tile (A rows) / column index (B/C N)
    const int half = lane >> 4;   // lane-group selector

    // ---- stage params to LDS ----
    for (int i = tid; i < 3*DDIM*RDIM; i += BLOCK) {
        int a = i / (DDIM*RDIM), rem = i % (DDIM*RDIM);
        int d = rem / RDIM,  r = rem % RDIM;
        sWdT[a][r][d] = (_Float16)(gamma[a*DDIM + d] * Wd[a*DDIM*RDIM + d*RDIM + r]);
    }
    for (int i = tid; i < 3*DDIM*KPAD; i += BLOCK) {
        int a = i / (DDIM*KPAD), rem = i % (DDIM*KPAD);
        int d = rem / KPAD, r = rem % KPAD;
        sWuT[a][d][r] = (r < RDIM) ? (_Float16)Wu[a*RDIM*DDIM + r*DDIM + d]
                                   : (_Float16)0.f;
    }
    for (int i = tid; i < 3*DDIM; i += BLOCK)
        sBu[i/DDIM][i%DDIM] = bu[i];
    for (int i = tid; i < 3*RDIM; i += BLOCK) {
        int a = i / RDIM, r = i % RDIM;
        float u = 0.f, v = 0.f;
        for (int d = 0; d < DDIM; ++d) {
            float w = Wd[a*DDIM*RDIM + d*RDIM + r];
            u += gamma[a*DDIM + d] * w;
            v += beta [a*DDIM + d] * w;
        }
        sU[a][r] = u;
        sV[a][r] = v + bd[a*RDIM + r];
    }
    __syncthreads();

    for (int tb = blockIdx.x; tb < nTileBlocks; tb += gridDim.x) {
        const int tile = tb * WAVES + wid;
        if (tile >= nTiles) continue;

        // ---- load 16 contiguous tokens (2048 floats), non-temporal, into xbuf ----
        const v4f* src = (const v4f*)xin + (size_t)tile * 512;
        v4f* xb4 = (v4f*)&xbuf[wid][0][0];
        #pragma unroll
        for (int j = 0; j < 16; ++j)
            xb4[j*32 + lane] = __builtin_nontemporal_load(&src[j*32 + lane]);

        // prefetch next tile for this wave (speculative; dropped if OOB)
        {
            int nt = tb + (int)gridDim.x;
            if (nt < nTileBlocks) {
                const float* p = xin + ((size_t)(nt*WAVES + wid)) * 2048 + lane*64;
                __builtin_prefetch(p, 0, 0);
            }
        }
        __threadfence_block();

        #pragma unroll
        for (int a = 0; a < 3; ++a) {
            // ---- LN stats: lanes (t, t+16) split token t's 128 dims ----
            v4f xr[16];
            {
                const v4f* xrp = (const v4f*)&xbuf[wid][t][half*64];
                #pragma unroll
                for (int i = 0; i < 16; ++i) xr[i] = xrp[i];
            }
            float s = 0.f, s2 = 0.f;
            #pragma unroll
            for (int i = 0; i < 16; ++i) {
                v4f v = xr[i];
                s  += v[0] + v[1] + v[2] + v[3];
                s2 += v[0]*v[0] + v[1]*v[1] + v[2]*v[2] + v[3]*v[3];
            }
            s  += __shfl_xor(s,  16, 32);
            s2 += __shfl_xor(s2, 16, 32);
            const float mu  = s * (1.0f/128.0f);
            const float var = s2 * (1.0f/128.0f) - mu*mu;
            const float rs  = rsqrtf(var + 1e-5f);
            if (half == 0) sStats[wid][t] = make_float2(rs, -mu * rs);

            // ---- raw x -> f16 A-feed (no per-element affine; folded into u/v) ----
            #pragma unroll
            for (int i8 = 0; i8 < 8; ++i8) {
                const v4f va = xr[i8*2], vb = xr[i8*2 + 1];
                v8h hv;
                hv[0] = (_Float16)va[0]; hv[1] = (_Float16)va[1];
                hv[2] = (_Float16)va[2]; hv[3] = (_Float16)va[3];
                hv[4] = (_Float16)vb[0]; hv[5] = (_Float16)vb[1];
                hv[6] = (_Float16)vb[2]; hv[7] = (_Float16)vb[3];
                *(v8h*)&hbuf[wid][t][half*64 + i8*8] = hv;
            }
            __threadfence_block();

            // ---- down-proj: 16x16 (tokens x R), K=128 in 4 chunks of 32 ----
            v8f acc = {};
            #pragma unroll
            for (int c = 0; c < 4; ++c) {
                // A layout (f16 16x32): lanes 0-15 K{0-7,16-23}, lanes 16-31 K{8-15,24-31}
                v8h alo = *(const v8h*)&hbuf[wid][t][c*32 + half*8];
                v8h ahi = *(const v8h*)&hbuf[wid][t][c*32 + half*8 + 16];
                v16h A;
                #pragma unroll
                for (int i = 0; i < 8; ++i) { A[i] = alo[i]; A[i+8] = ahi[i]; }
                // B layout: lanes 0-15 hold K{0-7,16-23} of column t; lanes 16-31 hold K{8-15,24-31}
                const v8h* wr = (const v8h*)&sWdT[a][t][c*32 + half*8];
                v8h b0 = wr[0], b1 = wr[2];
                v16h Bm;
                #pragma unroll
                for (int i = 0; i < 8; ++i) { Bm[i] = b0[i]; Bm[i+8] = b1[i]; }
                acc = __builtin_amdgcn_wmma_f32_16x16x32_f16(
                          false, A, false, Bm, (short)0, acc, false, false);
            }

            // ---- LN epilogue + exact GELU -> gbuf (C layout: N=t, M=half*8+j) ----
            {
                float2 st[8];
                const float2* stp = &sStats[wid][half*8];
                #pragma unroll
                for (int j = 0; j < 8; ++j) st[j] = stp[j];  // broadcast b128 loads
                const float uT = sU[a][t], vT = sV[a][t];
                #pragma unroll
                for (int j = 0; j < 8; ++j) {
                    float z  = fmaf(acc[j], st[j].x, fmaf(st[j].y, uT, vT));
                    float ge = 0.5f * z * (1.0f + erff(z * 0.70710678118654752f));
                    gbuf[wid][half*8 + j][t] = (_Float16)ge;
                }
            }
            __threadfence_block();

            // ---- up-proj: K=16 padded to 32 (B pad rows are stored zeros in LDS) ----
            v8h glo = *(const v8h*)&gbuf[wid][t][half*8];
            v16h Au;
            #pragma unroll
            for (int i = 0; i < 8; ++i) { Au[i] = glo[i]; Au[i+8] = (_Float16)0.f; }
            #pragma unroll
            for (int g = 0; g < 8; ++g) {
                const int dim = g*16 + t;
                const v8h* wp = (const v8h*)&sWuT[a][dim][half*8];
                v8h c0 = wp[0], c1 = wp[2];   // c1 reads the zero pad rows
                v16h Bu;
                #pragma unroll
                for (int i = 0; i < 8; ++i) { Bu[i] = c0[i]; Bu[i+8] = c1[i]; }
                v8f cu = {};
                cu = __builtin_amdgcn_wmma_f32_16x16x32_f16(
                         false, Au, false, Bu, (short)0, cu, false, false);
                const float buv = sBu[a][dim];
                #pragma unroll
                for (int j = 0; j < 8; ++j)
                    xbuf[wid][half*8 + j][dim] += cu[j] + buv;  // residual + bias
            }
            __threadfence_block();
        } // adapters

        // ---- store 16 tokens, coalesced, non-temporal ----
        v4f* dst = (v4f*)out + (size_t)tile * 512;
        #pragma unroll
        for (int j = 0; j < 16; ++j)
            __builtin_nontemporal_store(xb4[j*32 + lane], &dst[j*32 + lane]);
    }
}

extern "C" void kernel_launch(void* const* d_in, const int* in_sizes, int n_in,
                              void* d_out, int out_size, void* d_ws, size_t ws_size,
                              hipStream_t stream) {
    const float* x     = (const float*)d_in[0];
    const float* gamma = (const float*)d_in[1];
    const float* beta  = (const float*)d_in[2];
    const float* Wd    = (const float*)d_in[3];
    const float* bd    = (const float*)d_in[4];
    const float* Wu    = (const float*)d_in[5];
    const float* bu    = (const float*)d_in[6];
    float* out = (float*)d_out;

    const int nTokens     = in_sizes[0] / DDIM;
    const int nTiles      = nTokens / 16;
    const int nTileBlocks = (nTiles + WAVES - 1) / WAVES;
    const int blocks      = nTileBlocks < 4096 ? nTileBlocks : 4096;

    hipLaunchKernelGGL(adapter3_fused_wmma_kernel, dim3(blocks), dim3(BLOCK), 0, stream,
                       x, gamma, beta, Wd, bd, Wu, bu, out, nTiles, nTileBlocks);
}
